// MultiHeadAttention_with_preference_23725399343400
// MI455X (gfx1250) — matrix operand order repaired and checked by reading, not compiled, not run on hardware
//
#include <hip/hip_runtime.h>
#include <math.h>

// Problem constants (from reference): B=4, M=10, N=512, D=256, H=8, HD=32
#define B_  4
#define M_  10
#define N_  512
#define D_  256
#define H_  8
#define HD_ 32
#define S_  (M_ + N_)          /* 522 */
#define D3_ (3 * D_)           /* 768 */
#define SCALE_ 0.1767766952966369f  /* 1/sqrt(32) */

typedef __attribute__((ext_vector_type(16))) _Float16 v16h;
typedef __attribute__((ext_vector_type(4)))  _Float16 v4h;
typedef __attribute__((ext_vector_type(8)))  float    v8f;

// ---------------------------------------------------------------------------
// LDS-staged, double-buffered GEMM on the CDNA5 matrix pipe:
//   C[r,c] = (resid ? resid[r,c] : 0) + scale * act( sum_k A[r,k]*W[c,k] + bias[c] )
// A: [rows,K] f32 row-major, W: [cols,K] f32 row-major (C = A @ W^T).
// Block = 256 threads = 8 waves = 32x64 output tile; wave = one 16x16 WMMA tile.
// Tiles staged f32->f16 into LDS (padded stride 40 halves -> conflict-free
// 8-byte ds reads across the 16-lane groups), v_wmma_f32_16x16x32_f16 compute.
// ---------------------------------------------------------------------------
#define TM 32
#define TN 64
#define TK 32
#define LDSS 40   /* f16 elements per LDS row: 32 data + 8 pad (80B rows) */

__global__ __launch_bounds__(256)
void gemm_wmma_kernel(const float* __restrict__ A, const float* __restrict__ W,
                      const float* __restrict__ bias, const float* __restrict__ resid,
                      const float* __restrict__ scale, float* __restrict__ C,
                      int rows, int K, int cols, int relu)
{
    __shared__ _Float16 As[2][TM * LDSS];
    __shared__ _Float16 Bs[2][TN * LDSS];

    const int tid  = threadIdx.x;
    const int lane = tid & 31;
    const int wave = tid >> 5;
    const int blockRow = blockIdx.y * TM;
    const int blockCol = blockIdx.x * TN;
    const int half = lane >> 4;        // 0: lanes 0-15, 1: lanes 16-31
    const int l15  = lane & 15;

    // Cooperative staging indices: A tile 32x32 (4 f32/thread), B tile 64x32 (8 f32/thread)
    const int arow = tid >> 3;         // 0..31
    const int akq  = (tid & 7) * 4;    // 0..28 step 4
    const int wrow = tid >> 2;         // 0..63
    const int wkq  = (tid & 3) * 8;    // 0..24 step 8

    int aSrc = blockRow + arow; if (aSrc >= rows) aSrc = rows - 1;
    int wSrc = blockCol + wrow; if (wSrc >= cols) wSrc = cols - 1;
    const float* __restrict__ ap = A + (size_t)aSrc * K;
    const float* __restrict__ wp = W + (size_t)wSrc * K;

#define STAGE(kb_, p_) do {                                                         \
        const float4 a4  = *(const float4*)(ap + (kb_) + akq);                      \
        v4h ha; ha[0] = (_Float16)a4.x;  ha[1] = (_Float16)a4.y;                    \
                ha[2] = (_Float16)a4.z;  ha[3] = (_Float16)a4.w;                    \
        *(v4h*)(&As[p_][arow * LDSS + akq]) = ha;                                   \
        const float4 b4a = *(const float4*)(wp + (kb_) + wkq);                      \
        const float4 b4b = *(const float4*)(wp + (kb_) + wkq + 4);                  \
        v4h h0; h0[0] = (_Float16)b4a.x; h0[1] = (_Float16)b4a.y;                   \
                h0[2] = (_Float16)b4a.z; h0[3] = (_Float16)b4a.w;                   \
        v4h h1; h1[0] = (_Float16)b4b.x; h1[1] = (_Float16)b4b.y;                   \
                h1[2] = (_Float16)b4b.z; h1[3] = (_Float16)b4b.w;                   \
        *(v4h*)(&Bs[p_][wrow * LDSS + wkq])     = h0;                               \
        *(v4h*)(&Bs[p_][wrow * LDSS + wkq + 4]) = h1;                               \
    } while (0)

    // WMMA operand ISA layouts:
    //  A(16x32): lanes<16 hold K {0..7,16..23}; lanes>=16 hold {8..15,24..31}
    //  B(32x16): lane = n; lanes<16 hold K {0..7,16..23}; lanes>=16 hold {8..15,24..31}
    const int g    = half * 8;
    const int koff = half * 16;
    const _Float16* __restrict__ aLds = &As[0][0] + ((wave >> 2) * 16 + l15) * LDSS;
    const _Float16* __restrict__ bLds = &Bs[0][0] + ((wave & 3) * 16 + l15) * LDSS;
    const int bufStrideA = TM * LDSS;
    const int bufStrideB = TN * LDSS;

    v8f acc = {};
    const int nk = K / TK;

    STAGE(0, 0);
    __syncthreads();
    int p = 0;
    for (int t = 0; t < nk; ++t) {
        if (t + 1 < nk) STAGE((t + 1) * TK, p ^ 1);
        if (t + 2 < nk) {   // speculative prefetch of the tile after next
            __builtin_prefetch(ap + (t + 2) * TK + akq, 0, 1);
            __builtin_prefetch(wp + (t + 2) * TK + wkq, 0, 1);
        }
        const _Float16* aB = aLds + p * bufStrideA;
        const _Float16* bB = bLds + p * bufStrideB;
        v4h a0 = *(const v4h*)(aB + g);
        v4h a1 = *(const v4h*)(aB + g + 4);
        v4h a2 = *(const v4h*)(aB + 16 + g);
        v4h a3 = *(const v4h*)(aB + 16 + g + 4);
        v4h b0 = *(const v4h*)(bB + g);
        v4h b1 = *(const v4h*)(bB + g + 4);
        v4h b2 = *(const v4h*)(bB + 16 + g);
        v4h b3 = *(const v4h*)(bB + 16 + g + 4);
        v16h av, bv;
#pragma unroll
        for (int i = 0; i < 4; ++i) {
            av[i] = a0[i]; av[i + 4]  = a1[i]; av[i + 8]  = a2[i]; av[i + 12] = a3[i];
            bv[i] = b0[i]; bv[i + 4]  = b1[i]; bv[i + 8]  = b2[i]; bv[i + 12] = b3[i];
        }
        acc = __builtin_amdgcn_wmma_f32_16x16x32_f16(
                  false, av, false, bv, (short)0, acc, false, false);
        __syncthreads();
        p ^= 1;
    }
#undef STAGE

    // D layout: VGPR j -> M = j + 8*half ; N = lane&15
    const float sc  = scale ? scale[0] : 1.0f;
    const int   col = blockCol + (wave & 3) * 16 + l15;
    if (col < cols) {
        const float bc = bias ? bias[col] : 0.0f;
#pragma unroll
        for (int j = 0; j < 8; ++j) {
            int r = blockRow + (wave >> 2) * 16 + half * 8 + j;
            if (r < rows) {
                float v = acc[j] + bc;
                if (relu) v = fmaxf(v, 0.0f);
                v *= sc;
                if (resid) v += resid[(size_t)r * cols + col];
                C[(size_t)r * cols + col] = v;
            }
        }
    }
}

// ---------------------------------------------------------------------------
// Gather cust = node_emb[:, M:, :] into a contiguous [B*N, D] buffer (float4).
// ---------------------------------------------------------------------------
__global__ void copy_cust_kernel(const float* __restrict__ node, float* __restrict__ cust)
{
    int i = blockIdx.x * blockDim.x + threadIdx.x;          // over (B*N*D)/4
    if (i >= B_ * N_ * D_ / 4) return;
    int d4 = i % (D_ / 4);
    int r  = i / (D_ / 4);
    int b  = r / N_, n = r % N_;
    const float4* src = (const float4*)(node + ((size_t)(b * S_ + M_ + n)) * D_);
    ((float4*)cust)[i] = src[d4];
}

// ---------------------------------------------------------------------------
// Self-attention over customers (per b,h: 512x512), 2-pass online softmax,
// f32 VALU (flops negligible; softmax precision matters). float4 loads.
// qkv layout: [B*N, 3D]; q at +0, k at +D, v at +2D, head h at +h*HD.
// ---------------------------------------------------------------------------
__global__ __launch_bounds__(256)
void attn_self_kernel(const float* __restrict__ qkv, float* __restrict__ out)
{
    const int b = blockIdx.x / H_;
    const int h = blockIdx.x % H_;
    const int q = blockIdx.y * blockDim.x + threadIdx.x;    // 0..N_-1

    const float4* __restrict__ q4 =
        (const float4*)(qkv + ((size_t)(b * N_ + q)) * D3_ + h * HD_);
    float4 qr[HD_ / 4];
#pragma unroll
    for (int i = 0; i < HD_ / 4; ++i) qr[i] = q4[i];

    const float* __restrict__ kbase = qkv + ((size_t)(b * N_)) * D3_ + D_     + h * HD_;
    const float* __restrict__ vbase = qkv + ((size_t)(b * N_)) * D3_ + 2 * D_ + h * HD_;

    float m = -3.0e38f, l = 0.0f;
    for (int k = 0; k < N_; ++k) {
        const float4* kr = (const float4*)(kbase + (size_t)k * D3_);
        float s = 0.0f;
#pragma unroll
        for (int i = 0; i < HD_ / 4; ++i) {
            float4 kv = kr[i];
            s += qr[i].x * kv.x + qr[i].y * kv.y + qr[i].z * kv.z + qr[i].w * kv.w;
        }
        s *= SCALE_;
        float nm = fmaxf(m, s);
        l = l * __expf(m - nm) + __expf(s - nm);
        m = nm;
    }
    float4 accv[HD_ / 4];
#pragma unroll
    for (int i = 0; i < HD_ / 4; ++i) accv[i] = make_float4(0.f, 0.f, 0.f, 0.f);
    for (int k = 0; k < N_; ++k) {
        const float4* kr = (const float4*)(kbase + (size_t)k * D3_);
        float s = 0.0f;
#pragma unroll
        for (int i = 0; i < HD_ / 4; ++i) {
            float4 kv = kr[i];
            s += qr[i].x * kv.x + qr[i].y * kv.y + qr[i].z * kv.z + qr[i].w * kv.w;
        }
        float pw = __expf(s * SCALE_ - m);
        const float4* vr = (const float4*)(vbase + (size_t)k * D3_);
#pragma unroll
        for (int i = 0; i < HD_ / 4; ++i) {
            float4 vv = vr[i];
            accv[i].x += pw * vv.x; accv[i].y += pw * vv.y;
            accv[i].z += pw * vv.z; accv[i].w += pw * vv.w;
        }
    }
    const float inv = 1.0f / l;
    float4* o = (float4*)(out + ((size_t)(b * N_ + q)) * D_ + h * HD_);
#pragma unroll
    for (int i = 0; i < HD_ / 4; ++i)
        o[i] = make_float4(accv[i].x * inv, accv[i].y * inv, accv[i].z * inv, accv[i].w * inv);
}

// ---------------------------------------------------------------------------
// Cross attention: q from co (B,H,N,HD), k/v from ao (B,H,M,HD); softmax over M=10.
// ---------------------------------------------------------------------------
__global__ __launch_bounds__(256)
void attn_cross_kernel(const float* __restrict__ q2, const float* __restrict__ k2,
                       const float* __restrict__ v2, float* __restrict__ out)
{
    int idx = blockIdx.x * blockDim.x + threadIdx.x;        // B*H*N
    int b = idx / (H_ * N_);
    int rem = idx % (H_ * N_);
    int h = rem / N_;
    int n = rem % N_;

    const float4* __restrict__ q4 =
        (const float4*)(q2 + ((size_t)(b * N_ + n)) * D_ + h * HD_);
    float4 qr[HD_ / 4];
#pragma unroll
    for (int i = 0; i < HD_ / 4; ++i) qr[i] = q4[i];

    float s[M_];
    float m = -3.0e38f;
#pragma unroll
    for (int k = 0; k < M_; ++k) {
        const float4* kr = (const float4*)(k2 + ((size_t)(b * M_ + k)) * D_ + h * HD_);
        float t = 0.0f;
#pragma unroll
        for (int i = 0; i < HD_ / 4; ++i) {
            float4 kv = kr[i];
            t += qr[i].x * kv.x + qr[i].y * kv.y + qr[i].z * kv.z + qr[i].w * kv.w;
        }
        s[k] = t * SCALE_;
        m = fmaxf(m, s[k]);
    }
    float l = 0.0f;
#pragma unroll
    for (int k = 0; k < M_; ++k) { s[k] = __expf(s[k] - m); l += s[k]; }
    const float inv = 1.0f / l;

    float4 accv[HD_ / 4];
#pragma unroll
    for (int i = 0; i < HD_ / 4; ++i) accv[i] = make_float4(0.f, 0.f, 0.f, 0.f);
#pragma unroll
    for (int k = 0; k < M_; ++k) {
        const float4* vr = (const float4*)(v2 + ((size_t)(b * M_ + k)) * D_ + h * HD_);
#pragma unroll
        for (int i = 0; i < HD_ / 4; ++i) {
            float4 vv = vr[i];
            accv[i].x += s[k] * vv.x; accv[i].y += s[k] * vv.y;
            accv[i].z += s[k] * vv.z; accv[i].w += s[k] * vv.w;
        }
    }
    float4* o = (float4*)(out + ((size_t)(b * N_ + n)) * D_ + h * HD_);
#pragma unroll
    for (int i = 0; i < HD_ / 4; ++i)
        o[i] = make_float4(accv[i].x * inv, accv[i].y * inv, accv[i].z * inv, accv[i].w * inv);
}

// ---------------------------------------------------------------------------
// Degenerate preference-attention: softmax(axis=q).sum(axis=q) == 1, so
// ao_core[b,:] = sum_n cv[b,n,:]. Column-sum per batch (float4 columns).
// ---------------------------------------------------------------------------
__global__ void colsum_kernel(const float* __restrict__ cv, float* __restrict__ vsum)
{
    int i = blockIdx.x * blockDim.x + threadIdx.x;          // B*D/4
    if (i >= B_ * D_ / 4) return;
    int b = i / (D_ / 4), d4 = i % (D_ / 4);
    const float4* p = (const float4*)(cv + (size_t)b * N_ * D_) + d4;
    float4 s = make_float4(0.f, 0.f, 0.f, 0.f);
    for (int n = 0; n < N_; ++n) {
        float4 v = p[(size_t)n * (D_ / 4)];
        s.x += v.x; s.y += v.y; s.z += v.z; s.w += v.w;
    }
    ((float4*)vsum)[i] = s;
}

// ao[b,m,:] = agent[b,m,:] + t[b,:] * a2   (t identical for all m)
__global__ void assemble_ao_kernel(const float* __restrict__ node, const float* __restrict__ t,
                                   const float* __restrict__ a2, float* __restrict__ ao)
{
    int i = blockIdx.x * blockDim.x + threadIdx.x;          // B*M*D/4
    if (i >= B_ * M_ * D_ / 4) return;
    int d4 = i % (D_ / 4);
    int r  = i / (D_ / 4);
    int b  = r / M_, m = r % M_;
    float4 agent = ((const float4*)(node + ((size_t)(b * S_ + m)) * D_))[d4];
    float4 tv    = ((const float4*)(t + b * D_))[d4];
    float a = a2[0];
    ((float4*)ao)[i] = make_float4(agent.x + tv.x * a, agent.y + tv.y * a,
                                   agent.z + tv.z * a, agent.w + tv.w * a);
}

// out[:, :M] = ao ; out[:, M:] = co2
__global__ void final_copy_kernel(const float* __restrict__ ao, const float* __restrict__ co2,
                                  float* __restrict__ out)
{
    int i = blockIdx.x * blockDim.x + threadIdx.x;          // B*S*D/4
    if (i >= B_ * S_ * D_ / 4) return;
    int d4 = i % (D_ / 4);
    int r  = i / (D_ / 4);
    int b  = r / S_, s = r % S_;
    float4 v = (s < M_)
        ? ((const float4*)(ao  + ((size_t)(b * M_ + s)) * D_))[d4]
        : ((const float4*)(co2 + ((size_t)(b * N_ + (s - M_))) * D_))[d4];
    ((float4*)out)[i] = v;
}

// ---------------------------------------------------------------------------
// Host orchestration
// ---------------------------------------------------------------------------
extern "C" void kernel_launch(void* const* d_in, const int* in_sizes, int n_in,
                              void* d_out, int out_size, void* d_ws, size_t ws_size,
                              hipStream_t stream)
{
    // Flattened input order (setup_inputs insertion order, params nested in order):
    // 0 node_emb, 1 prefer_emb,
    // 2/3 wqkv.w/b, 4/5 wq1, 6/7 wk1, 8/9 wv1, 10/11 wq2, 12/13 wk2, 14/15 wv2,
    // 16/17 out_navi, 18/19 out1, 20/21 out2, 22/23 wq_p, 24/25 wk_p,
    // 26 a1, 27 a2, 28 a3,
    // 29-33 ff1{l1.w,l1.b,l2.w,l2.b,alpha}, 34-38 ff2{...}, 39-43 ff3{...}
    const float* node   = (const float*)d_in[0];
    // d_in[1] (prefer_emb), wq1/wk1/wq_p/wk_p are provably dead: softmax over q
    // summed over q == 1, so attn == ones and the whole path cancels.
    const float* wqkv_w = (const float*)d_in[2];  const float* wqkv_b = (const float*)d_in[3];
    const float* wv1_w  = (const float*)d_in[8];  const float* wv1_b  = (const float*)d_in[9];
    const float* wq2_w  = (const float*)d_in[10]; const float* wq2_b  = (const float*)d_in[11];
    const float* wk2_w  = (const float*)d_in[12]; const float* wk2_b  = (const float*)d_in[13];
    const float* wv2_w  = (const float*)d_in[14]; const float* wv2_b  = (const float*)d_in[15];
    const float* onv_w  = (const float*)d_in[16]; const float* onv_b  = (const float*)d_in[17];
    const float* o1_w   = (const float*)d_in[18]; const float* o1_b   = (const float*)d_in[19];
    const float* o2_w   = (const float*)d_in[20]; const float* o2_b   = (const float*)d_in[21];
    const float* a1     = (const float*)d_in[26];
    const float* a2     = (const float*)d_in[27];
    const float* a3     = (const float*)d_in[28];
    const float* f1l1w  = (const float*)d_in[29]; const float* f1l1b = (const float*)d_in[30];
    const float* f1l2w  = (const float*)d_in[31]; const float* f1l2b = (const float*)d_in[32];
    const float* f1al   = (const float*)d_in[33];
    const float* f2l1w  = (const float*)d_in[34]; const float* f2l1b = (const float*)d_in[35];
    const float* f2l2w  = (const float*)d_in[36]; const float* f2l2b = (const float*)d_in[37];
    const float* f2al   = (const float*)d_in[38];
    const float* f3l1w  = (const float*)d_in[39]; const float* f3l1b = (const float*)d_in[40];
    const float* f3l2w  = (const float*)d_in[41]; const float* f3l2b = (const float*)d_in[42];
    const float* f3al   = (const float*)d_in[43];
    float* out = (float*)d_out;

    const int RN = B_ * N_;        // 2048 customer rows
    const int RM = B_ * M_;        // 40 agent rows
    float* ws = (float*)d_ws;
    size_t o = 0;
    float* cust  = ws + o; o += (size_t)RN * D_;
    float* qkv   = ws + o; o += (size_t)RN * D3_;
    float* atnA  = ws + o; o += (size_t)RN * D_;
    float* coA   = ws + o; o += (size_t)RN * D_;
    float* ffh   = ws + o; o += (size_t)RN * D3_;   // shared hidden buffer (ff1/ff2/ff3)
    float* coB   = ws + o; o += (size_t)RN * D_;    // "co"
    float* cv    = ws + o; o += (size_t)RN * D_;
    float* vsum  = ws + o; o += (size_t)B_ * D_;
    float* tbuf  = ws + o; o += (size_t)B_ * D_;
    float* aoA   = ws + o; o += (size_t)RM * D_;
    float* aoB   = ws + o; o += (size_t)RM * D_;    // "ao"
    float* q2    = ws + o; o += (size_t)RN * D_;
    float* k2    = ws + o; o += (size_t)RM * D_;
    float* v2    = ws + o; o += (size_t)RM * D_;
    float* atnB  = ws + o; o += (size_t)RN * D_;
    float* co2A  = ws + o; o += (size_t)RN * D_;
    float* co2B  = ws + o; o += (size_t)RN * D_;
    (void)ws_size; (void)n_in; (void)in_sizes; (void)out_size;

    auto gemm = [&](const float* A, const float* W, const float* bias, const float* resid,
                    const float* scale, float* C, int rows, int K, int cols, int relu) {
        dim3 g((cols + TN - 1) / TN, (rows + TM - 1) / TM);
        gemm_wmma_kernel<<<g, dim3(256), 0, stream>>>(A, W, bias, resid, scale, C,
                                                      rows, K, cols, relu);
    };

    // 1) cust = node_emb[:, M:, :]
    copy_cust_kernel<<<(RN * D_ / 4 + 255) / 256, 256, 0, stream>>>(node, cust);
    // 2) qkv = cust @ Wqkv^T + b
    gemm(cust, wqkv_w, wqkv_b, nullptr, nullptr, qkv, RN, D_, D3_, 0);
    // 3) self-attention over customers
    attn_self_kernel<<<dim3(B_ * H_, N_ / 256), 256, 0, stream>>>(qkv, atnA);
    // 4) co = cust + lin(attn, out_navi) * a1
    gemm(atnA, onv_w, onv_b, cust, a1, coA, RN, D_, D_, 0);
    // 5-6) co = ff1(co)
    gemm(coA, f1l1w, f1l1b, nullptr, nullptr, ffh, RN, D_, D3_, 1);
    gemm(ffh,  f1l2w, f1l2b, coA,     f1al,    coB, RN, D3_, D_, 0);
    // 7) cv = co @ wv1^T + b ; 8) vsum[b,:] = sum_n cv[b,n,:]  (attn == ones)
    gemm(coB, wv1_w, wv1_b, nullptr, nullptr, cv, RN, D_, D_, 0);
    colsum_kernel<<<(B_ * D_ / 4 + 255) / 256, 256, 0, stream>>>(cv, vsum);
    // 9) t = vsum @ out1^T + b ; 10) ao = agent + t * a2
    gemm(vsum, o1_w, o1_b, nullptr, nullptr, tbuf, B_, D_, D_, 0);
    assemble_ao_kernel<<<(RM * D_ / 4 + 255) / 256, 256, 0, stream>>>(node, tbuf, a2, aoA);
    // 11-12) ao = ff2(ao)
    gemm(aoA, f2l1w, f2l1b, nullptr, nullptr, ffh, RM, D_, D3_, 1);
    gemm(ffh, f2l2w, f2l2b, aoA,     f2al,    aoB, RM, D3_, D_, 0);
    // 13-15) q2 from co, k2/v2 from ao
    gemm(coB, wq2_w, wq2_b, nullptr, nullptr, q2, RN, D_, D_, 0);
    gemm(aoB, wk2_w, wk2_b, nullptr, nullptr, k2, RM, D_, D_, 0);
    gemm(aoB, wv2_w, wv2_b, nullptr, nullptr, v2, RM, D_, D_, 0);
    // 16) cross attention (softmax over M=10 keys)
    attn_cross_kernel<<<(B_ * H_ * N_) / 256, 256, 0, stream>>>(q2, k2, v2, atnB);
    // 17) co2 = co + lin(attnB, out2) * a3
    gemm(atnB, o2_w, o2_b, coB, a3, co2A, RN, D_, D_, 0);
    // 18-19) co2 = ff3(co2)
    gemm(co2A, f3l1w, f3l1b, nullptr, nullptr, ffh,  RN, D_, D3_, 1);
    gemm(ffh,  f3l2w, f3l2b, co2A,    f3al,    co2B, RN, D3_, D_, 0);
    // 20) out = concat([ao, co2], axis=1)
    final_copy_kernel<<<(B_ * S_ * D_ / 4 + 255) / 256, 256, 0, stream>>>(aoB, co2B, out);
}
